// ExpertMLP_29678224016194
// MI455X (gfx1250) — 8-bit matrix operand order repaired and checked by reading, compiled, not run on hardware
//
#include <hip/hip_runtime.h>
#include <stdint.h>

typedef __attribute__((ext_vector_type(8))) int v8i;

#define D_MODEL 2048
#define D_FF    8192
#define TOKENS  (4 * 4096)

#define KSTEP   64
#define LDS_PITCH 80
#define BLK_M   64
#define BLK_N   256

__global__ __launch_bounds__(256) void abs_sum_partial_kernel(
    const float* __restrict__ w, long long n, float* __restrict__ partials) {
  __shared__ float red[256];
  float s = 0.f;
  long long stride = (long long)gridDim.x * blockDim.x;
  for (long long i = (long long)blockIdx.x * blockDim.x + threadIdx.x; i < n; i += stride)
    s += fabsf(w[i]);
  red[threadIdx.x] = s;
  __syncthreads();
  for (int off = 128; off > 0; off >>= 1) {
    if ((int)threadIdx.x < off) red[threadIdx.x] += red[threadIdx.x + off];
    __syncthreads();
  }
  if (threadIdx.x == 0) partials[blockIdx.x] = red[0];
}

__global__ __launch_bounds__(256) void reduce_partials_kernel(
    const float* __restrict__ partials, int n, float* __restrict__ out) {
  __shared__ float red[256];
  float s = 0.f;
  for (int i = threadIdx.x; i < n; i += 256) s += partials[i];
  red[threadIdx.x] = s;
  __syncthreads();
  for (int off = 128; off > 0; off >>= 1) {
    if ((int)threadIdx.x < off) red[threadIdx.x] += red[threadIdx.x + off];
    __syncthreads();
  }
  if (threadIdx.x == 0) *out = red[0];
}

__global__ __launch_bounds__(256) void weight_quant_kernel(
    const float* __restrict__ w, long long n, const float* __restrict__ sum,
    float inv_n, int8_t* __restrict__ wq, float* __restrict__ deq_out) {
  float mean = fmaxf(*sum * inv_n, 1e-5f);
  float scale = 1.f / mean;
  long long stride = (long long)gridDim.x * blockDim.x;
  for (long long i = (long long)blockIdx.x * blockDim.x + threadIdx.x; i < n; i += stride) {
    float q = rintf(w[i] * scale);
    q = fminf(fmaxf(q, -1.f), 1.f);
    wq[i] = (int8_t)q;
  }
  if (blockIdx.x == 0 && threadIdx.x == 0) *deq_out = mean;
}

__global__ __launch_bounds__(256) void act_quant_kernel(
    const float* __restrict__ x, int d, int8_t* __restrict__ xq,
    float* __restrict__ deq) {
  __shared__ float red[256];
  long long row = blockIdx.x;
  const float4* xr = (const float4*)(x + row * (long long)d);
  int nv = d >> 2;
  float amax = 0.f;
  for (int i = threadIdx.x; i < nv; i += 256) {
    float4 v = xr[i];
    amax = fmaxf(amax, fmaxf(fmaxf(fabsf(v.x), fabsf(v.y)),
                             fmaxf(fabsf(v.z), fabsf(v.w))));
  }
  red[threadIdx.x] = amax;
  __syncthreads();
  for (int off = 128; off > 0; off >>= 1) {
    if ((int)threadIdx.x < off)
      red[threadIdx.x] = fmaxf(red[threadIdx.x], red[threadIdx.x + off]);
    __syncthreads();
  }
  float a = fmaxf(red[0], 1e-5f);
  float scale = 127.f / a;
  if (threadIdx.x == 0) deq[row] = a * (1.f / 127.f);
  char4* q = (char4*)(xq + row * (long long)d);
  for (int i = threadIdx.x; i < nv; i += 256) {
    float4 v = xr[i];
    char4 c;
    c.x = (signed char)fminf(fmaxf(rintf(v.x * scale), -128.f), 127.f);
    c.y = (signed char)fminf(fmaxf(rintf(v.y * scale), -128.f), 127.f);
    c.z = (signed char)fminf(fmaxf(rintf(v.z * scale), -128.f), 127.f);
    c.w = (signed char)fminf(fmaxf(rintf(v.w * scale), -128.f), 127.f);
    q[i] = c;
  }
}

__device__ __forceinline__ v8i lds_a_frag(const int8_t* p) {
  const uint2* ap = (const uint2*)p;
  uint2 a0 = ap[0], a1 = ap[1], a2 = ap[4], a3 = ap[5];
  v8i a;
  a[0] = (int)a0.x; a[1] = (int)a0.y;
  a[2] = (int)a1.x; a[3] = (int)a1.y;
  a[4] = (int)a2.x; a[5] = (int)a2.y;
  a[6] = (int)a3.x; a[7] = (int)a3.y;
  return a;
}

__device__ __forceinline__ v8i lds_b_frag(const int8_t* p) {
  const int4* bp = (const int4*)p;
  int4 b0 = bp[0], b1 = bp[2];
  v8i b;
  b[0] = b0.x; b[1] = b0.y; b[2] = b0.z; b[3] = b0.w;
  b[4] = b1.x; b[5] = b1.y; b[6] = b1.z; b[7] = b1.w;
  return b;
}

#define WMMA_IU8(a, b, c) \
  __builtin_amdgcn_wmma_i32_16x16x64_iu8(true, (a), true, (b), (c), false, false)

__device__ __forceinline__ unsigned lds_off(const void* p) {
  return (unsigned)(uintptr_t)p;
}

__device__ __forceinline__ void async_stage(
    const int8_t* __restrict__ Ag, const int8_t* __restrict__ Bg, int K, int k,
    int8_t* ash, int8_t* bsh, int tid, int rowBase, int colBase) {
  {
    const int r = tid >> 3, o = (tid & 7) * 8;
    const int8_t* g = Ag + (size_t)(rowBase + r) * K + k + o;
    unsigned l = lds_off(ash + r * LDS_PITCH + o);
    asm volatile("global_load_async_to_lds_b64 %0, %1, off"
                 :: "v"(l), "v"((unsigned long long)(uintptr_t)g) : "memory");
  }
  {
    const int r = tid >> 1, o = (tid & 1) * 32;
    const int8_t* g = Bg + (size_t)(colBase + r) * K + k + o;
    unsigned l = lds_off(bsh + r * LDS_PITCH + o);
    asm volatile("global_load_async_to_lds_b128 %0, %1, off"
                 :: "v"(l), "v"((unsigned long long)(uintptr_t)g) : "memory");
    asm volatile("global_load_async_to_lds_b128 %0, %1, off offset:16"
                 :: "v"(l), "v"((unsigned long long)(uintptr_t)g) : "memory");
  }
}

__global__ __launch_bounds__(512) void bitgemm_wmma_kernel(
    const int8_t* __restrict__ A, const int8_t* __restrict__ Bw,
    float* __restrict__ C, int M, int N, int K,
    const float* __restrict__ a_deq, const float* __restrict__ w_deq,
    int do_silu) {
  __shared__ int8_t Ash[3][BLK_M * LDS_PITCH];
  __shared__ int8_t Bsh[3][BLK_N * LDS_PITCH];

  const int tid  = threadIdx.x;
  const int lane = tid & 31;
  const int wave = tid >> 5;
  const int mw   = wave & 1;
  const int ng   = wave >> 1;
  const int rowBase = blockIdx.y * BLK_M;
  const int colBase = blockIdx.x * BLK_N;
  const int ml   = lane & 15;
  const int half = lane >> 4;

  const int aoff0 = (mw * 32 + ml) * LDS_PITCH + half * 16;
  const int aoff1 = (mw * 32 + 16 + ml) * LDS_PITCH + half * 16;
  const int boff0 = (ng * 32 + ml) * LDS_PITCH + half * 16;
  const int boff1 = (ng * 32 + 16 + ml) * LDS_PITCH + half * 16;

  v8i acc0 = {}, acc1 = {}, acc2 = {}, acc3 = {};

  const int nk = K >> 6;

  async_stage(A, Bw, K, 0, Ash[0], Bsh[0], tid, rowBase, colBase);
  async_stage(A, Bw, K, (nk > 1 ? KSTEP : 0), Ash[1], Bsh[1], tid, rowBase, colBase);

#pragma unroll 1
  for (int i = 0; i < nk; ++i) {
    asm volatile("s_wait_asynccnt 0x3" ::: "memory");
    __syncthreads();

    const int kn = (i + 2 < nk) ? (i + 2) * KSTEP : i * KSTEP;
    const int nbuf = (i + 2) % 3;
    async_stage(A, Bw, K, kn, Ash[nbuf], Bsh[nbuf], tid, rowBase, colBase);

    const int8_t* ab = Ash[i % 3];
    const int8_t* bb = Bsh[i % 3];
    v8i a0 = lds_a_frag(ab + aoff0);
    v8i a1 = lds_a_frag(ab + aoff1);
    v8i b0 = lds_b_frag(bb + boff0);
    v8i b1 = lds_b_frag(bb + boff1);

    acc0 = WMMA_IU8(a0, b0, acc0);
    acc1 = WMMA_IU8(a0, b1, acc1);
    acc2 = WMMA_IU8(a1, b0, acc2);
    acc3 = WMMA_IU8(a1, b1, acc3);
  }

  const float wd = *w_deq;
  const v8i accs[4] = {acc0, acc1, acc2, acc3};
#pragma unroll
  for (int mi = 0; mi < 2; ++mi) {
#pragma unroll
    for (int ni = 0; ni < 2; ++ni) {
      const v8i a8 = accs[mi * 2 + ni];
      const int col = colBase + ng * 32 + ni * 16 + ml;
#pragma unroll
      for (int r = 0; r < 8; ++r) {
        const int row = rowBase + mw * 32 + mi * 16 + half * 8 + r;
        float v = (float)a8[r] * a_deq[row] * wd;
        if (do_silu) v = v * (1.f / (1.f + __expf(-v)));
        C[(size_t)row * N + col] = v;
      }
    }
  }
}

extern "C" void kernel_launch(void* const* d_in, const int* in_sizes, int n_in,
                              void* d_out, int out_size, void* d_ws,
                              size_t ws_size, hipStream_t stream) {
  (void)in_sizes; (void)n_in; (void)out_size; (void)ws_size;
  const float* x  = (const float*)d_in[0];
  const float* W1 = (const float*)d_in[1];
  const float* W2 = (const float*)d_in[2];
  float* out = (float*)d_out;

  char* ws = (char*)d_ws;
  size_t off = 0;
  int8_t* w1q = (int8_t*)(ws + off); off += (size_t)D_FF * D_MODEL;
  int8_t* w2q = (int8_t*)(ws + off); off += (size_t)D_MODEL * D_FF;
  int8_t* xq  = (int8_t*)(ws + off); off += (size_t)TOKENS * D_MODEL;
  int8_t* hq  = (int8_t*)(ws + off); off += (size_t)TOKENS * D_FF;
  off = (off + 255) & ~(size_t)255;
  float* h    = (float*)(ws + off);  off += (size_t)TOKENS * D_FF * 4;
  float* xdeq = (float*)(ws + off);  off += (size_t)TOKENS * 4;
  float* hdeq = (float*)(ws + off);  off += (size_t)TOKENS * 4;
  float* part = (float*)(ws + off);  off += 2048 * 4;
  float* scal = (float*)(ws + off);  off += 4 * 4;

  const long long n1 = (long long)D_FF * D_MODEL;
  const long long n2 = (long long)D_MODEL * D_FF;

  abs_sum_partial_kernel<<<2048, 256, 0, stream>>>(W1, n1, part);
  reduce_partials_kernel<<<1, 256, 0, stream>>>(part, 2048, &scal[0]);
  weight_quant_kernel<<<2048, 256, 0, stream>>>(W1, n1, &scal[0],
                                                1.f / (float)n1, w1q, &scal[2]);
  abs_sum_partial_kernel<<<2048, 256, 0, stream>>>(W2, n2, part);
  reduce_partials_kernel<<<1, 256, 0, stream>>>(part, 2048, &scal[1]);
  weight_quant_kernel<<<2048, 256, 0, stream>>>(W2, n2, &scal[1],
                                                1.f / (float)n2, w2q, &scal[3]);

  act_quant_kernel<<<TOKENS, 256, 0, stream>>>(x, D_MODEL, xq, xdeq);

  bitgemm_wmma_kernel<<<dim3(D_FF / BLK_N, TOKENS / BLK_M), 512, 0, stream>>>(
      xq, w1q, h, TOKENS, D_FF, D_MODEL, xdeq, &scal[2], 1);

  act_quant_kernel<<<TOKENS, 256, 0, stream>>>(h, D_FF, hq, hdeq);

  bitgemm_wmma_kernel<<<dim3(D_MODEL / BLK_N, TOKENS / BLK_M), 512, 0, stream>>>(
      hq, w2q, out, TOKENS, D_MODEL, D_FF, hdeq, &scal[3], 0);
}
